// DualStreamGNNEncoder_27582279975547
// MI455X (gfx1250) — compile-verified, matrix operand order checked by reading, not run on hardware
//
#include <hip/hip_runtime.h>
#include <hip/hip_bf16.h>
#include <math.h>

#define Bk 16
#define Nk 512
#define Hk 128
#define NNk (Nk*Nk)
#define NEGB (-1e9f)

typedef __bf16 bf16;
typedef __attribute__((ext_vector_type(16))) __bf16 v16bf;
typedef __attribute__((ext_vector_type(8)))  __bf16 v8bf;
typedef __attribute__((ext_vector_type(8)))  float  v8f;
typedef __attribute__((ext_vector_type(2)))  float  v2f;

// ---------------------------------------------------------------- fragments
// bf16 A/B fragment (16x32 A or row of Bt): lanes<16 K=[k0..k0+8)+[k0+16..k0+24),
// lanes>=16 K=[k0+8..k0+16)+[k0+24..k0+32)
__device__ __forceinline__ v16bf ldfrag(const bf16* __restrict__ base, int ld, int r, int k0, int hi) {
  const bf16* p = base + (size_t)r * ld + k0 + (hi << 3);
  v8bf lo = *(const v8bf*)p;
  v8bf hi8 = *(const v8bf*)(p + 16);
  v16bf f;
#pragma unroll
  for (int e = 0; e < 8; ++e) { f[e] = lo[e]; f[8 + e] = hi8[e]; }
  return f;
}

template<int ACT> __device__ __forceinline__ float actf(float v) {
  if (ACT == 1) return fmaxf(v, 0.f);
  if (ACT == 2) return 0.5f * v * (1.f + erff(v * 0.70710678118654752f));
  return v;
}

// ---------------------------------------------------------------- generic bf16 WMMA GEMM
// D[M,N] = act(A[M,K] @ Bt[N,K]^T + bias), batched via grid.z with element strides.
// Optional bf16 mirror Dbf (bfT=1 writes transposed n-major for later use as Bt).
// K-loop is software-pipelined: fragments for step i+1 are issued before the
// WMMAs of step i, so the compiler can emit partial s_wait_loadcnt and overlap
// VMEM latency with the matrix pipe.
template<int ACT>
__global__ __launch_bounds__(256)
void k_gemm(const bf16* __restrict__ A, const bf16* __restrict__ Bt,
            const float* __restrict__ bias,
            float* __restrict__ D, bf16* __restrict__ Dbf,
            int M, int N, int K, int lda, int ldb, int ldd,
            long sA, long sB, long sD,
            int ldbf, int bfT, long sBf, int bfColOff)
{
  const int bz = blockIdx.z;
  const bf16* Ab = A + (size_t)bz * sA;
  const bf16* Bb = Bt + (size_t)bz * sB;
  const int tid = threadIdx.x, lane = tid & 31, wv = tid >> 5;
  const int hi = lane >> 4, l15 = lane & 15;
  const int m0 = blockIdx.x * 128 + wv * 16;
  const int n0 = blockIdx.y * 64;
  if (m0 >= M) return;
  const v8f z = {0.f,0.f,0.f,0.f,0.f,0.f,0.f,0.f};
  v8f acc[4] = {z, z, z, z};
  const int arow = m0 + l15;
  const int br0 = n0 + 0 * 16 + l15, br1 = n0 + 1 * 16 + l15;
  const int br2 = n0 + 2 * 16 + l15, br3 = n0 + 3 * 16 + l15;

  // prologue: fragments for k0 = 0
  v16bf af = ldfrag(Ab, lda, arow, 0, hi);
  v16bf b0 = ldfrag(Bb, ldb, br0, 0, hi);
  v16bf b1 = ldfrag(Bb, ldb, br1, 0, hi);
  v16bf b2 = ldfrag(Bb, ldb, br2, 0, hi);
  v16bf b3 = ldfrag(Bb, ldb, br3, 0, hi);

  for (int k0 = 32; k0 < K; k0 += 32) {
    // issue next-step loads first (overlap with current WMMAs)
    v16bf afn = ldfrag(Ab, lda, arow, k0, hi);
    v16bf b0n = ldfrag(Bb, ldb, br0, k0, hi);
    v16bf b1n = ldfrag(Bb, ldb, br1, k0, hi);
    v16bf b2n = ldfrag(Bb, ldb, br2, k0, hi);
    v16bf b3n = ldfrag(Bb, ldb, br3, k0, hi);
    acc[0] = __builtin_amdgcn_wmma_f32_16x16x32_bf16(false, af, false, b0, (short)0, acc[0], false, false);
    acc[1] = __builtin_amdgcn_wmma_f32_16x16x32_bf16(false, af, false, b1, (short)0, acc[1], false, false);
    acc[2] = __builtin_amdgcn_wmma_f32_16x16x32_bf16(false, af, false, b2, (short)0, acc[2], false, false);
    acc[3] = __builtin_amdgcn_wmma_f32_16x16x32_bf16(false, af, false, b3, (short)0, acc[3], false, false);
    af = afn; b0 = b0n; b1 = b1n; b2 = b2n; b3 = b3n;
  }
  // epilogue WMMA group
  acc[0] = __builtin_amdgcn_wmma_f32_16x16x32_bf16(false, af, false, b0, (short)0, acc[0], false, false);
  acc[1] = __builtin_amdgcn_wmma_f32_16x16x32_bf16(false, af, false, b1, (short)0, acc[1], false, false);
  acc[2] = __builtin_amdgcn_wmma_f32_16x16x32_bf16(false, af, false, b2, (short)0, acc[2], false, false);
  acc[3] = __builtin_amdgcn_wmma_f32_16x16x32_bf16(false, af, false, b3, (short)0, acc[3], false, false);

#pragma unroll
  for (int j = 0; j < 4; ++j) {
    int n = n0 + j * 16 + l15;
#pragma unroll
    for (int r = 0; r < 8; ++r) {
      int m = m0 + r + 8 * hi;
      if (m < M && n < N) {
        float v = acc[j][r];
        if (bias) v += bias[n];
        v = actf<ACT>(v);
        if (D) D[(size_t)bz * sD + (size_t)m * ldd + n] = v;
        if (Dbf) {
          if (bfT) Dbf[(size_t)bz * sBf + (size_t)n * ldbf + m] = (bf16)v;
          else     Dbf[(size_t)bz * sBf + (size_t)m * ldbf + bfColOff + n] = (bf16)v;
        }
      }
    }
  }
}

// ---------------------------------------------------------------- cross-lane reductions (16-lane groups)
__device__ __forceinline__ float rmax16(float v) {
  v = fmaxf(v, __shfl_xor(v, 1, 32));
  v = fmaxf(v, __shfl_xor(v, 2, 32));
  v = fmaxf(v, __shfl_xor(v, 4, 32));
  v = fmaxf(v, __shfl_xor(v, 8, 32));
  return v;
}
__device__ __forceinline__ float rsum16(float v) {
  v += __shfl_xor(v, 1, 32);
  v += __shfl_xor(v, 2, 32);
  v += __shfl_xor(v, 4, 32);
  v += __shfl_xor(v, 8, 32);
  return v;
}

// ---------------------------------------------------------------- fused flash attention
// One wave per 16-row q tile. S and P@V via v_wmma_f32_16x16x4_f32.
// MODE 0: dist-bin bias + key mask (phys); MODE 1: key mask only (fusion).
template<int DH, int MODE>
__global__ __launch_bounds__(128)
void k_attn(const float* __restrict__ Q, const float* __restrict__ Km, const float* __restrict__ Vm,
            int qld, int kld, int vld,
            const unsigned char* __restrict__ bins, const float* __restrict__ dbias,
            const unsigned char* __restrict__ kmask,
            float* __restrict__ O, bf16* __restrict__ Obf, int oldd, float scale)
{
  __shared__ float ldsP[4 * 256];
  const int lane = threadIdx.x & 31, wv = threadIdx.x >> 5;
  const int hi = lane >> 4, l15 = lane & 15;
  const int h = blockIdx.y, bb = blockIdx.z;
  const int qt = blockIdx.x * 4 + wv;
  const int hoff = h * DH;
  const v8f z = {0.f,0.f,0.f,0.f,0.f,0.f,0.f,0.f};

  // Q fragment registers (A layout, fixed all k-loop)
  const float* qr = Q + ((size_t)bb * Nk + qt * 16 + l15) * qld + hoff;
  float qa[DH / 2];
#pragma unroll
  for (int s = 0; s < DH / 4; ++s) {
    qa[2 * s]     = qr[4 * s + 2 * hi];
    qa[2 * s + 1] = qr[4 * s + 2 * hi + 1];
  }

  float mrow[8], lrow[8];
#pragma unroll
  for (int r = 0; r < 8; ++r) { mrow[r] = -1e30f; lrow[r] = 0.f; }
  v8f Oa[DH / 16];
#pragma unroll
  for (int j = 0; j < DH / 16; ++j) Oa[j] = z;

  float* myP = ldsP + wv * 256;
  const unsigned char* binb = (MODE == 0) ? (bins + (size_t)bb * NNk) : nullptr;

  for (int kt = 0; kt < Nk / 16; ++kt) {
    // --- scores S = Q K^T (16x16), reduction over DH in steps of 4
    v8f Sa = z;
    const float* kr = Km + ((size_t)bb * Nk + kt * 16 + l15) * kld + hoff;
#pragma unroll
    for (int s = 0; s < DH / 4; ++s) {
      v2f a = {qa[2 * s], qa[2 * s + 1]};
      v2f b = {kr[4 * s + 2 * hi], kr[4 * s + 2 * hi + 1]};
      Sa = __builtin_amdgcn_wmma_f32_16x16x4_f32(false, a, false, b, (short)0, Sa, false, false);
    }
    const int key = kt * 16 + l15;
    float kb = 0.f;
    if (kmask && kmask[(size_t)bb * Nk + key] == 0) kb = NEGB;

    float pv[8];
#pragma unroll
    for (int r = 0; r < 8; ++r) {
      int qrow = qt * 16 + r + 8 * hi;
      float sv = Sa[r] * scale + kb;
      if (MODE == 0) sv += dbias[h * 16 + binb[(size_t)qrow * Nk + key]];
      float rm = rmax16(sv);
      float mnew = fmaxf(mrow[r], rm);
      float alpha = expf(mrow[r] - mnew);
      float p = expf(sv - mnew);
      float rs = rsum16(p);
      lrow[r] = lrow[r] * alpha + rs;
      mrow[r] = mnew;
      pv[r] = p;
#pragma unroll
      for (int j = 0; j < DH / 16; ++j) Oa[j][r] *= alpha;
    }
    // stage P to LDS (plain row-major 16x16 per wave) to re-enter A-frag layout
#pragma unroll
    for (int r = 0; r < 8; ++r) myP[(r + 8 * hi) * 16 + l15] = pv[r];
    __asm__ volatile("" ::: "memory");
    // --- O += P @ Vtile  (reduction over 16 keys in steps of 4)
    const float* vbase = Vm + ((size_t)bb * Nk) * vld + hoff;
#pragma unroll
    for (int s = 0; s < 4; ++s) {
      v2f pa = {myP[l15 * 16 + 4 * s + 2 * hi], myP[l15 * 16 + 4 * s + 2 * hi + 1]};
      int krow = kt * 16 + 4 * s + 2 * hi;
#pragma unroll
      for (int j = 0; j < DH / 16; ++j) {
        v2f vb = {vbase[(size_t)krow * vld + j * 16 + l15],
                  vbase[(size_t)(krow + 1) * vld + j * 16 + l15]};
        Oa[j] = __builtin_amdgcn_wmma_f32_16x16x4_f32(false, pa, false, vb, (short)0, Oa[j], false, false);
      }
    }
  }
  // epilogue: O /= l
#pragma unroll
  for (int r = 0; r < 8; ++r) {
    int qrow = qt * 16 + r + 8 * hi;
    float inv = lrow[r] > 0.f ? 1.f / lrow[r] : 0.f;
#pragma unroll
    for (int j = 0; j < DH / 16; ++j) {
      float v = Oa[j][r] * inv;
      size_t oidx = ((size_t)bb * Nk + qrow) * oldd + hoff + j * 16 + l15;
      O[oidx] = v;
      if (Obf) Obf[oidx] = (bf16)v;
    }
  }
}

// ---------------------------------------------------------------- LayerNorm variants (row = 128)
__device__ __forceinline__ void blk_reduce2(float& s1, float& s2) {
  __shared__ float a[128], c[128];
  int t = threadIdx.x;
  a[t] = s1; c[t] = s2; __syncthreads();
  for (int o = 64; o > 0; o >>= 1) {
    if (t < o) { a[t] += a[t + o]; c[t] += c[t + o]; }
    __syncthreads();
  }
  s1 = a[0]; s2 = c[0];
}

// MODE 0: y=LN(x); MODE 1: y=LN(x+res); MODE 2: y=relu(LN(x))+res
template<int MODE>
__global__ __launch_bounds__(128)
void k_ln(const float* __restrict__ x, const float* __restrict__ res,
          const float* __restrict__ g, const float* __restrict__ b,
          float* __restrict__ y, bf16* __restrict__ ybf, int ldbf, int coloff)
{
  int row = blockIdx.x, t = threadIdx.x;
  size_t idx = (size_t)row * Hk + t;
  float v = x[idx];
  if (MODE == 1) v += res[idx];
  float s1 = v, s2 = v * v;
  blk_reduce2(s1, s2);
  float mean = s1 / Hk, var = s2 / Hk - mean * mean;
  float rstd = rsqrtf(var + 1e-5f);
  float o = (v - mean) * rstd * g[t] + b[t];
  if (MODE == 2) o = fmaxf(o, 0.f) + res[idx];
  if (y) y[idx] = o;
  if (ybf) ybf[(size_t)row * ldbf + coloff + t] = (bf16)o;
}

// input projection: LN(relu(x @ w + bias))   (w is (KIN,128) row-major, tiny K)
template<int KIN>
__global__ __launch_bounds__(128)
void k_inproj(const float* __restrict__ x, const float* __restrict__ w, const float* __restrict__ bias,
              const float* __restrict__ g, const float* __restrict__ bln,
              float* __restrict__ y, bf16* __restrict__ ybf)
{
  int row = blockIdx.x, t = threadIdx.x;
  float acc = bias[t];
#pragma unroll
  for (int k = 0; k < KIN; ++k) acc += x[(size_t)row * KIN + k] * w[k * Hk + t];
  acc = fmaxf(acc, 0.f);
  float s1 = acc, s2 = acc * acc;
  blk_reduce2(s1, s2);
  float mean = s1 / Hk, var = s2 / Hk - mean * mean;
  float rstd = rsqrtf(var + 1e-5f);
  float o = (acc - mean) * rstd * g[t] + bln[t];
  y[(size_t)row * Hk + t] = o;
  ybf[(size_t)row * Hk + t] = (bf16)o;
}

// ---------------------------------------------------------------- small prep kernels
template<int RELU, int MASKED>
__global__ void k_smax(const float* __restrict__ x, const unsigned char* __restrict__ m,
                       float* __restrict__ mx)
{
  __shared__ float sm[256];
  int b = blockIdx.x;
  const float* xb = x + (size_t)b * NNk;
  float best = 0.f;
  for (int i = threadIdx.x; i < NNk; i += 256) {
    float v = xb[i];
    if (RELU) v = fmaxf(v, 0.f);
    if (MASKED) {
      int r = i / Nk, c = i % Nk;
      if (!(m[b * Nk + r] && m[b * Nk + c])) v = 0.f;
    }
    best = fmaxf(best, v);
  }
  sm[threadIdx.x] = best; __syncthreads();
  for (int o = 128; o > 0; o >>= 1) {
    if (threadIdx.x < o) sm[threadIdx.x] = fmaxf(sm[threadIdx.x], sm[threadIdx.x + o]);
    __syncthreads();
  }
  if (threadIdx.x == 0) mx[b] = fmaxf(sm[0], 1e-6f);
}

__global__ void k_bins(const float* __restrict__ d, const float* __restrict__ mx,
                       unsigned char* __restrict__ bins)
{
  size_t id = (size_t)blockIdx.x * blockDim.x + threadIdx.x;
  if (id >= (size_t)Bk * NNk) return;
  int b = (int)(id / NNk);
  int v = (int)(d[id] / mx[b] * 15.0f);
  v = v < 0 ? 0 : (v > 15 ? 15 : v);
  bins[id] = (unsigned char)v;
}

__global__ void k_flow_adj(const float* __restrict__ fl, const unsigned char* __restrict__ m,
                           const float* __restrict__ mx, float* __restrict__ A2)
{
  size_t id = (size_t)blockIdx.x * blockDim.x + threadIdx.x;
  if (id >= (size_t)Bk * NNk) return;
  int b = (int)(id / NNk); int rem = (int)(id % NNk);
  int i = rem / Nk, j = rem % Nk;
  float v = fl[id]; v = v > 0.f ? v : 0.f;
  float mi = m[b * Nk + i] ? 1.f : 0.f, mj = m[b * Nk + j] ? 1.f : 0.f;
  v = v * mi * mj / mx[b];
  if (i == j) v += mi;
  A2[id] = v;
}

__global__ void k_colsum(const float* __restrict__ A2, float* __restrict__ dinv)
{
  __shared__ float sm[256];
  int j = blockIdx.x, b = blockIdx.y;
  float s = 0.f;
  for (int i = threadIdx.x; i < Nk; i += 256) s += A2[(size_t)b * NNk + (size_t)i * Nk + j];
  sm[threadIdx.x] = s; __syncthreads();
  for (int o = 128; o > 0; o >>= 1) {
    if (threadIdx.x < o) sm[threadIdx.x] += sm[threadIdx.x + o];
    __syncthreads();
  }
  if (threadIdx.x == 0) { float d = sm[0]; dinv[b * Nk + j] = d > 0.f ? rsqrtf(d) : 0.f; }
}

// AnT[b][j][i] = dinv_i * A2[i][j] * dinv_j  (bf16, ready as GEMM A operand)
__global__ void k_an_t(const float* __restrict__ A2, const float* __restrict__ dinv,
                       bf16* __restrict__ AnT)
{
  size_t id = (size_t)blockIdx.x * blockDim.x + threadIdx.x;
  if (id >= (size_t)Bk * NNk) return;
  int b = (int)(id / NNk); int rem = (int)(id % NNk);
  int j = rem / Nk, i = rem % Nk;
  float an = dinv[b * Nk + i] * A2[(size_t)b * NNk + (size_t)i * Nk + j] * dinv[b * Nk + j];
  AnT[id] = (bf16)an;
}

__global__ __launch_bounds__(128)
void k_gather(const float* __restrict__ src, const int* __restrict__ idx,
              float* __restrict__ y, bf16* __restrict__ ybf)
{
  int row = blockIdx.x, t = threadIdx.x;
  int b = row / Nk;
  int d = idx[row]; d = d < 0 ? 0 : (d >= Nk ? Nk - 1 : d);
  float v = src[((size_t)b * Nk + d) * Hk + t];
  y[(size_t)row * Hk + t] = v;
  ybf[(size_t)row * Hk + t] = (bf16)v;
}

__global__ __launch_bounds__(128)
void k_maskmul(const float* __restrict__ x, const unsigned char* __restrict__ m,
               bf16* __restrict__ ybf)
{
  int row = blockIdx.x, t = threadIdx.x;
  float mm = m[row] ? 1.f : 0.f;
  ybf[(size_t)row * Hk + t] = (bf16)(x[(size_t)row * Hk + t] * mm);
}

// weight prep: transpose (K,N)->(N,K) + convert to bf16
__global__ void k_wt(const float* __restrict__ w, bf16* __restrict__ wt, int K, int N)
{
  int id = blockIdx.x * blockDim.x + threadIdx.x;
  if (id >= K * N) return;
  int n = id / K, k = id % K;
  wt[id] = (bf16)w[(size_t)k * N + n];
}
__global__ void k_wc(const float* __restrict__ w, bf16* __restrict__ o, int n)
{
  int id = blockIdx.x * blockDim.x + threadIdx.x;
  if (id < n) o[id] = (bf16)w[id];
}

// ================================================================ host
extern "C" void kernel_launch(void* const* d_in, const int* in_sizes, int n_in,
                              void* d_out, int out_size, void* d_ws, size_t ws_size,
                              hipStream_t stream)
{
  auto F  = [&](int i) { return (const float*)d_in[i]; };
  auto U8 = [&](int i) { return (const unsigned char*)d_in[i]; };
  auto I32= [&](int i) { return (const int*)d_in[i]; };

  char* ws = (char*)d_ws;
  size_t off = 0;
  auto alloc = [&](size_t bytes) -> char* {
    off = (off + 255) & ~(size_t)255;
    char* p = ws + off; off += bytes; return p;
  };

  // ---- param indices (setup_inputs() pytree, depth-first insertion order)
  const int P_PHYS_INP_W = 9, P_PHYS_INP_B = 10, P_PHYS_LN_G = 11, P_PHYS_LN_B = 12, P_DIST_BIAS = 13;
  auto PL = [](int l, int k) { return 14 + 12 * l + k; }; // qkv.w,b out.w,b ff1.w,b ff2.w,b ln1.g,b ln2.g,b
  const int P_PHYS_OUTP_W = 62, P_PHYS_OUTP_B = 63;
  const int P_FLOW_INP_W = 64, P_FLOW_INP_B = 65, P_FLOW_LN_G = 66, P_FLOW_LN_B = 67;
  auto FL = [](int l, int k) { return 68 + 4 * l + k; }; // gcn.w,b ln.g,b
  const int P_FLOW_OUTP_W = 80, P_FLOW_OUTP_B = 81;
  const int P_P2F_INW = 82, P_P2F_INB = 83, P_P2F_OUTW = 84, P_P2F_OUTB = 85;
  const int P_F2P_INW = 86, P_F2P_INB = 87, P_F2P_OUTW = 88, P_F2P_OUTB = 89;
  const int P_LNP_G = 90, P_LNP_B = 91, P_LNF_G = 92, P_LNF_B = 93;
  const int P_MLP1_W = 94, P_MLP1_B = 95, P_MLP2_W = 96, P_MLP2_B = 97, P_MLPLN_G = 98, P_MLPLN_B = 99;

  // ---- convert weights to bf16 (lin weights (K,N) -> transposed (N,K); in_w already (N,K))
  auto make_wt = [&](int widx, int K, int N) -> bf16* {
    bf16* p = (bf16*)alloc((size_t)K * N * 2);
    int tot = K * N;
    k_wt<<<(tot + 255) / 256, 256, 0, stream>>>(F(widx), p, K, N);
    return p;
  };
  auto make_wc = [&](int widx, int n) -> bf16* {
    bf16* p = (bf16*)alloc((size_t)n * 2);
    k_wc<<<(n + 255) / 256, 256, 0, stream>>>(F(widx), p, n);
    return p;
  };

  bf16 *Wqkv[4], *Wout[4], *Wff1[4], *Wff2[4];
  for (int l = 0; l < 4; ++l) {
    Wqkv[l] = make_wt(PL(l, 0), 128, 384);
    Wout[l] = make_wt(PL(l, 2), 128, 128);
    Wff1[l] = make_wt(PL(l, 4), 128, 512);
    Wff2[l] = make_wt(PL(l, 6), 512, 128);
  }
  bf16* Wpout = make_wt(P_PHYS_OUTP_W, 128, 128);
  bf16* Wgcn[3];
  for (int l = 0; l < 3; ++l) Wgcn[l] = make_wt(FL(l, 0), 128, 128);
  bf16* Wfout = make_wt(P_FLOW_OUTP_W, 128, 128);
  bf16* Wp2f_in  = make_wc(P_P2F_INW, 384 * 128);
  bf16* Wp2f_out = make_wt(P_P2F_OUTW, 128, 128);
  bf16* Wf2p_in  = make_wc(P_F2P_INW, 384 * 128);
  bf16* Wf2p_out = make_wt(P_F2P_OUTW, 128, 128);
  bf16* Wmlp1 = make_wt(P_MLP1_W, 256, 256);
  bf16* Wmlp2 = make_wt(P_MLP2_W, 256, 128);

  // ---- activation buffers
  const size_t BN = (size_t)Bk * Nk; // 8192
  float* h_phys    = (float*)alloc(BN * Hk * 4);
  bf16*  h_phys_bf = (bf16*)alloc(BN * Hk * 2);
  float* qkv       = (float*)alloc(BN * 3 * Hk * 4);
  float* attn      = (float*)alloc(BN * Hk * 4);
  bf16*  attn_bf   = (bf16*)alloc(BN * Hk * 2);
  float* gbuf      = (float*)alloc(BN * Hk * 4);
  bf16*  tmid_bf   = (bf16*)alloc(BN * 4 * Hk * 2);
  unsigned char* bins = (unsigned char*)alloc((size_t)Bk * NNk);
  float* mxd  = (float*)alloc(Bk * 4);
  float* mxf  = (float*)alloc(Bk * 4);
  float* A2   = (float*)alloc((size_t)Bk * NNk * 4);
  float* dinv = (float*)alloc(BN * 4);
  bf16*  AnT  = (bf16*)alloc((size_t)Bk * NNk * 2);
  float* h_flow    = (float*)alloc(BN * Hk * 4);
  bf16*  h_flow_bf = (bf16*)alloc(BN * Hk * 2);
  bf16*  tmpT      = (bf16*)alloc((size_t)Bk * Hk * Nk * 2);
  float* hpe    = (float*)alloc(BN * Hk * 4);
  bf16*  hpe_bf = (bf16*)alloc(BN * Hk * 2);
  float* hfe    = (float*)alloc(BN * Hk * 4);
  bf16*  hfe_bf = (bf16*)alloc(BN * Hk * 2);
  float* h_fa    = (float*)alloc(BN * Hk * 4);
  bf16*  h_fa_bf = (bf16*)alloc(BN * Hk * 2);
  bf16*  hc = (bf16*)alloc(BN * 2 * Hk * 2);

  auto gemm = [&](int ACT, const bf16* A, const bf16* Bt, const float* bias,
                  float* D, bf16* Dbf, int M, int N, int K, int lda, int ldb, int ldd,
                  long sA, long sB, long sD, int ldbf, int bfT, long sBf, int coloff, int batch) {
    dim3 g((M + 127) / 128, N / 64, batch), blk(256);
    if (ACT == 0)      k_gemm<0><<<g, blk, 0, stream>>>(A, Bt, bias, D, Dbf, M, N, K, lda, ldb, ldd, sA, sB, sD, ldbf, bfT, sBf, coloff);
    else if (ACT == 1) k_gemm<1><<<g, blk, 0, stream>>>(A, Bt, bias, D, Dbf, M, N, K, lda, ldb, ldd, sA, sB, sD, ldbf, bfT, sBf, coloff);
    else               k_gemm<2><<<g, blk, 0, stream>>>(A, Bt, bias, D, Dbf, M, N, K, lda, ldb, ldd, sA, sB, sD, ldbf, bfT, sBf, coloff);
  };

  // ================= physical encoder =================
  k_inproj<4><<<(int)BN, 128, 0, stream>>>(F(0), F(P_PHYS_INP_W), F(P_PHYS_INP_B),
                                           F(P_PHYS_LN_G), F(P_PHYS_LN_B), h_phys, h_phys_bf);
  k_smax<0, 0><<<Bk, 256, 0, stream>>>(F(1), nullptr, mxd);
  {
    size_t tot = (size_t)Bk * NNk;
    k_bins<<<(unsigned)((tot + 255) / 256), 256, 0, stream>>>(F(1), mxd, bins);
  }
  for (int l = 0; l < 4; ++l) {
    gemm(0, h_phys_bf, Wqkv[l], F(PL(l, 1)), qkv, nullptr, 8192, 384, 128, 128, 128, 384, 0, 0, 0, 0, 0, 0, 0, 1);
    dim3 ga(Nk / 64, 8, Bk);
    k_attn<16, 0><<<ga, 128, 0, stream>>>(qkv + 0, qkv + 128, qkv + 256, 384, 384, 384,
                                          bins, F(P_DIST_BIAS), U8(2), attn, attn_bf, 128, 0.25f);
    gemm(0, attn_bf, Wout[l], F(PL(l, 3)), gbuf, nullptr, 8192, 128, 128, 128, 128, 128, 0, 0, 0, 0, 0, 0, 0, 1);
    k_ln<1><<<(int)BN, 128, 0, stream>>>(gbuf, h_phys, F(PL(l, 8)), F(PL(l, 9)), h_phys, h_phys_bf, 128, 0);
    gemm(2, h_phys_bf, Wff1[l], F(PL(l, 5)), nullptr, tmid_bf, 8192, 512, 128, 128, 128, 512, 0, 0, 0, 512, 0, 0, 0, 1);
    gemm(0, tmid_bf, Wff2[l], F(PL(l, 7)), gbuf, nullptr, 8192, 128, 512, 512, 512, 128, 0, 0, 0, 0, 0, 0, 0, 1);
    k_ln<1><<<(int)BN, 128, 0, stream>>>(gbuf, h_phys, F(PL(l, 10)), F(PL(l, 11)), h_phys, h_phys_bf, 128, 0);
  }
  gemm(0, h_phys_bf, Wpout, F(P_PHYS_OUTP_B), hpe, hpe_bf, 8192, 128, 128, 128, 128, 128, 0, 0, 0, 128, 0, 0, 0, 1);

  // ================= flow encoder (GCN) =================
  k_inproj<2><<<(int)BN, 128, 0, stream>>>(F(3), F(P_FLOW_INP_W), F(P_FLOW_INP_B),
                                           F(P_FLOW_LN_G), F(P_FLOW_LN_B), h_flow, h_flow_bf);
  k_smax<1, 1><<<Bk, 256, 0, stream>>>(F(4), U8(5), mxf);
  {
    size_t tot = (size_t)Bk * NNk;
    unsigned nb = (unsigned)((tot + 255) / 256);
    k_flow_adj<<<nb, 256, 0, stream>>>(F(4), U8(5), mxf, A2);
    k_colsum<<<dim3(Nk, Bk), 256, 0, stream>>>(A2, dinv);
    k_an_t<<<nb, 256, 0, stream>>>(A2, dinv, AnT);
  }
  for (int l = 0; l < 3; ++l) {
    // tmpT[b] = (h W)^T  (128 x 512, bf16)  via transposed mirror
    gemm(0, h_flow_bf, Wgcn[l], nullptr, nullptr, tmpT, 512, 128, 128, 128, 128, 128,
         (long)512 * 128, 0, 0, 512, 1, (long)Hk * Nk, 0, Bk);
    // g[b] = AnT[b] (512x512) @ tmpT[b]^T + bias
    gemm(0, AnT, tmpT, F(FL(l, 1)), gbuf, nullptr, 512, 128, 512, 512, 512, 128,
         (long)NNk, (long)Hk * Nk, (long)512 * 128, 0, 0, 0, 0, Bk);
    k_ln<2><<<(int)BN, 128, 0, stream>>>(gbuf, h_flow, F(FL(l, 2)), F(FL(l, 3)), h_flow, h_flow_bf, 128, 0);
  }
  k_maskmul<<<(int)BN, 128, 0, stream>>>(h_flow, U8(5), h_flow_bf);
  gemm(0, h_flow_bf, Wfout, F(P_FLOW_OUTP_B), hfe, hfe_bf, 8192, 128, 128, 128, 128, 128, 0, 0, 0, 128, 0, 0, 0, 1);

  // ================= fusion =================
  k_gather<<<(int)BN, 128, 0, stream>>>(hfe, I32(7), h_fa, h_fa_bf);
  float* qf = qkv;
  float* kf = qkv + BN * Hk;
  float* vf = qkv + 2 * BN * Hk;
  const float sc32 = 0.17677669529663687f;
  dim3 gaf(Nk / 64, 4, Bk);

  // p2f: q = h_phys_enc, k = v = h_fa
  gemm(0, hpe_bf,  Wp2f_in,              F(P_P2F_INB),       qf, nullptr, 8192, 128, 128, 128, 128, 128, 0, 0, 0, 0, 0, 0, 0, 1);
  gemm(0, h_fa_bf, Wp2f_in + 128 * 128,  F(P_P2F_INB) + 128, kf, nullptr, 8192, 128, 128, 128, 128, 128, 0, 0, 0, 0, 0, 0, 0, 1);
  gemm(0, h_fa_bf, Wp2f_in + 2 * 128 * 128, F(P_P2F_INB) + 256, vf, nullptr, 8192, 128, 128, 128, 128, 128, 0, 0, 0, 0, 0, 0, 0, 1);
  k_attn<32, 1><<<gaf, 128, 0, stream>>>(qf, kf, vf, 128, 128, 128, nullptr, nullptr, U8(8),
                                         attn, attn_bf, 128, sc32);
  gemm(0, attn_bf, Wp2f_out, F(P_P2F_OUTB), gbuf, nullptr, 8192, 128, 128, 128, 128, 128, 0, 0, 0, 0, 0, 0, 0, 1);
  k_ln<1><<<(int)BN, 128, 0, stream>>>(gbuf, hpe, F(P_LNP_G), F(P_LNP_B), nullptr, hc, 256, 0);

  // f2p: q = h_fa, k = v = h_phys_enc
  gemm(0, h_fa_bf, Wf2p_in,              F(P_F2P_INB),       qf, nullptr, 8192, 128, 128, 128, 128, 128, 0, 0, 0, 0, 0, 0, 0, 1);
  gemm(0, hpe_bf,  Wf2p_in + 128 * 128,  F(P_F2P_INB) + 128, kf, nullptr, 8192, 128, 128, 128, 128, 128, 0, 0, 0, 0, 0, 0, 0, 1);
  gemm(0, hpe_bf,  Wf2p_in + 2 * 128 * 128, F(P_F2P_INB) + 256, vf, nullptr, 8192, 128, 128, 128, 128, 128, 0, 0, 0, 0, 0, 0, 0, 1);
  k_attn<32, 1><<<gaf, 128, 0, stream>>>(qf, kf, vf, 128, 128, 128, nullptr, nullptr, U8(8),
                                         attn, attn_bf, 128, sc32);
  gemm(0, attn_bf, Wf2p_out, F(P_F2P_OUTB), gbuf, nullptr, 8192, 128, 128, 128, 128, 128, 0, 0, 0, 0, 0, 0, 0, 1);
  k_ln<1><<<(int)BN, 128, 0, stream>>>(gbuf, h_fa, F(P_LNF_G), F(P_LNF_B), nullptr, hc, 256, 128);

  // MLP on concat + final LN -> d_out
  gemm(2, hc, Wmlp1, F(P_MLP1_B), nullptr, tmid_bf, 8192, 256, 256, 256, 256, 256, 0, 0, 0, 256, 0, 0, 0, 1);
  gemm(0, tmid_bf, Wmlp2, F(P_MLP2_B), gbuf, nullptr, 8192, 128, 256, 256, 256, 128, 0, 0, 0, 0, 0, 0, 0, 1);
  k_ln<0><<<(int)BN, 128, 0, stream>>>(gbuf, nullptr, F(P_MLPLN_G), F(P_MLPLN_B), (float*)d_out, nullptr, 0, 0);

  (void)in_sizes; (void)n_in; (void)out_size; (void)ws_size;
}
